// MambaAudioFeatureExtractor_86045374808330
// MI455X (gfx1250) — matrix operand order repaired and checked by reading, not compiled, not run on hardware
//
#include <hip/hip_runtime.h>
#include <hip/hip_bf16.h>

typedef __attribute__((ext_vector_type(16))) _Float16 v16h;
typedef __attribute__((ext_vector_type(8)))  float    v8f;
typedef __attribute__((ext_vector_type(4)))  unsigned v4u;
typedef __attribute__((ext_vector_type(8)))  unsigned v8u;

#define SELU_A 1.6732632423543772f
#define SELU_S 1.0507009873554805f

static __device__ __forceinline__ float softplusf(float x) {
    return (x > 20.f) ? x : __logf(1.f + __expf(x));
}
static __device__ __forceinline__ float siluf(float x) {
    return x / (1.f + __expf(-x));
}
static __device__ __forceinline__ float seluf(float x) {
    return x > 0.f ? SELU_S * x : SELU_S * SELU_A * (__expf(x) - 1.f);
}

// ---------------------------------------------------------------------------
// Wave-level 16x16 output tile, f16 inputs, f32 accumulate.
// A tile: 16 rows x K (row-major, ldA elems). B tile: 16 tokens x K (row-major,
// ldB elems) -> serves as the (K x 16) B operand with N = token-in-tile.
// Fragment layouts per CDNA5 ISA 7.12.2 (wave32).
// ---------------------------------------------------------------------------
template <int K>
static __device__ __forceinline__ v8f wmma_tile(const _Float16* __restrict__ At, int ldA,
                                                const _Float16* __restrict__ Bt, int ldB,
                                                v8f acc, int lane) {
    const int r  = lane & 15;
    const int hi = lane >> 4;
#pragma unroll
    for (int k0 = 0; k0 < K; k0 += 32) {
        v16h a, b;
#pragma unroll
        for (int e = 0; e < 8; ++e) {
            a[e]     = At[r * ldA + k0 + hi * 8 + e];          // K = hi*8 + e
            a[e + 8] = At[r * ldA + k0 + hi * 8 + 16 + e];     // K = hi*8 + 16 + e
        }
#pragma unroll
        for (int e = 0; e < 8; ++e) {
            b[e]     = Bt[r * ldB + k0 + hi * 8 + e];          // N=r, K = hi*8 + e
            b[e + 8] = Bt[r * ldB + k0 + hi * 8 + 16 + e];     // N=r, K = hi*8 + 16 + e
        }
        acc = __builtin_amdgcn_wmma_f32_16x16x32_f16(false, a, false, b,
                                                     (short)0, acc, false, false);
    }
    return acc;
}

// ---------------------------------------------------------------------------
// Weight prep: f32 -> f16 (W_x zero-padded 36->48 rows), Aneg = -exp(A_log)
// ---------------------------------------------------------------------------
__global__ void prep_layer_kernel(const float* __restrict__ Win, const float* __restrict__ Wx,
                                  const float* __restrict__ Wout, const float* __restrict__ Alog,
                                  _Float16* __restrict__ WinH, _Float16* __restrict__ WxH,
                                  _Float16* __restrict__ WoutH, float* __restrict__ Aneg) {
    int i = blockIdx.x * blockDim.x + threadIdx.x;          // 64*256 = 16384
    if (i < 256 * 64) WinH[i] = (_Float16)Win[i];
    if (i < 48 * 128) {
        int r = i >> 7;
        WxH[i] = (r < 36) ? (_Float16)Wx[i] : (_Float16)0.f;
    }
    if (i < 64 * 128) WoutH[i] = (_Float16)Wout[i];
    if (i < 128 * 16) Aneg[i] = -__expf(Alog[i]);
}

__global__ void prep_wf_kernel(const float* __restrict__ Wf, _Float16* __restrict__ WfH) {
    int i = blockIdx.x * blockDim.x + threadIdx.x;
    if (i < 64 * 64) WfH[i] = (_Float16)Wf[i];
}

// ---------------------------------------------------------------------------
// Frontend: conv1(s=64,k=128,p=32)+bn+selu -> conv2(k=3,p=1)+bn+selu
//           -> conv3(k=3,p=1)+bn+selu.  One block = (batch, 64-timestep tile).
// ---------------------------------------------------------------------------
__global__ void frontend_kernel(const float* __restrict__ x,
    const float* __restrict__ w1, const float* __restrict__ b1,
    const float* __restrict__ g1, const float* __restrict__ bb1,
    const float* __restrict__ m1, const float* __restrict__ v1,
    const float* __restrict__ w2, const float* __restrict__ b2,
    const float* __restrict__ g2, const float* __restrict__ bb2,
    const float* __restrict__ m2, const float* __restrict__ v2,
    const float* __restrict__ w3, const float* __restrict__ b3,
    const float* __restrict__ g3, const float* __restrict__ bb3,
    const float* __restrict__ m3, const float* __restrict__ v3,
    float* __restrict__ y, _Float16* __restrict__ yh) {
    __shared__ float xin[4416];        // samples for 68 conv1 outputs
    __shared__ float c1[68 * 32];
    __shared__ float c2[66 * 64];
    const int tid  = threadIdx.x;      // 256
    const int b    = blockIdx.y;
    const int t0   = blockIdx.x * 64;
    const int s_lo = t0 - 2;
    const long lo  = (long)s_lo * 64 - 32;
    const float* xb = x + (size_t)b * 262144;

    for (int i = tid; i < 4416; i += 256) {
        long m = lo + i;
        xin[i] = (m >= 0 && m < 262144) ? xb[m] : 0.f;
    }
    __syncthreads();
    for (int i = tid; i < 68 * 32; i += 256) {
        int srow = i >> 5, oc = i & 31;
        int s = s_lo + srow;
        float val = 0.f;
        if (s >= 0 && s < 4096) {
            float acc = b1[oc];
            const float* wr = w1 + oc * 128;
            const int base = srow * 64;
#pragma unroll 8
            for (int k = 0; k < 128; ++k) acc += wr[k] * xin[base + k];
            val = seluf(g1[oc] * (acc - m1[oc]) * rsqrtf(v1[oc] + 1e-5f) + bb1[oc]);
        }
        c1[srow * 32 + oc] = val;
    }
    __syncthreads();
    for (int i = tid; i < 66 * 64; i += 256) {
        int srow = i >> 6, oc = i & 63;
        int s = t0 - 1 + srow;
        float val = 0.f;
        if (s >= 0 && s < 4096) {
            float acc = b2[oc];
            const float* wr = w2 + oc * 96;
#pragma unroll
            for (int k = 0; k < 3; ++k)
                for (int ic = 0; ic < 32; ++ic)
                    acc += wr[ic * 3 + k] * c1[(srow + k) * 32 + ic];
            val = seluf(g2[oc] * (acc - m2[oc]) * rsqrtf(v2[oc] + 1e-5f) + bb2[oc]);
        }
        c2[srow * 64 + oc] = val;
    }
    __syncthreads();
    for (int i = tid; i < 64 * 64; i += 256) {
        int srow = i >> 6, oc = i & 63;
        float acc = b3[oc];
        const float* wr = w3 + oc * 192;
#pragma unroll
        for (int k = 0; k < 3; ++k)
            for (int ic = 0; ic < 64; ++ic)
                acc += wr[ic * 3 + k] * c2[(srow + k) * 64 + ic];
        float val = seluf(g3[oc] * (acc - m3[oc]) * rsqrtf(v3[oc] + 1e-5f) + bb3[oc]);
        size_t tok = (size_t)b * 4096 + t0 + srow;
        y[tok * 64 + oc]  = val;
        yh[tok * 64 + oc] = (_Float16)val;
    }
}

// ---------------------------------------------------------------------------
// GEMM: xz[tok, 0:256] = W_in(256x64) @ y[tok, 0:64]   (WMMA, K=64)
// B tile (2KB) via per-lane async global->LDS copy (ASYNCcnt), issued first;
// W_in (32KB) via a single TDM tensor_load_to_lds 1-D descriptor (TENSORcnt)
// issued by wave 0 -- copies overlap, vector-memory pipe stays free.
// ---------------------------------------------------------------------------
__global__ void gemm_in_kernel(const _Float16* __restrict__ yh,
                               const _Float16* __restrict__ WinH,
                               float* __restrict__ xz, int ntok) {
    __shared__ _Float16 Bt[16 * 64];
    __shared__ _Float16 At[256 * 64];
    const int tid  = threadIdx.x;         // 128
    const int tok0 = blockIdx.x * 16;

    {   // activation tile: 2KB, one b128 async lane-copy per thread
        unsigned lofs = (unsigned)(unsigned long long)(const void*)Bt + (unsigned)tid * 16u;
        unsigned long long gsrc =
            (unsigned long long)(const void*)(yh + (size_t)tok0 * 64) +
            (unsigned long long)tid * 16ull;
        asm volatile("global_load_async_to_lds_b128 %0, %1, off"
                     :: "v"(lofs), "v"(gsrc) : "memory");
        if (tok0 + 16 < ntok)
            __builtin_prefetch(yh + (size_t)(tok0 + 16) * 64, 0, 1);
    }
    if (tid < 32) {   // wave 0: TDM 1-D copy of W_in (4096 x 8B elements)
        unsigned long long ga = (unsigned long long)(const void*)WinH;
        unsigned ldsA = (unsigned)(unsigned long long)(const void*)At;
        v4u g0;
        g0[0] = 1u;                                               // count=1
        g0[1] = ldsA;                                             // lds_addr (bytes)
        g0[2] = (unsigned)ga;                                     // global_addr[31:0]
        g0[3] = (unsigned)((ga >> 32) & 0x1FFFFFFull) | (2u << 30); // addr[56:32], type=2
        v8u g1;
        g1[0] = 3u << 16;                // data_size = 8B, no multicast/pad/iterate
        g1[1] = (4096u & 0xFFFFu) << 16; // tensor_dim0[15:0] = 4096
        g1[2] = (4096u >> 16) | (1u << 16); // tensor_dim0[31:16], tensor_dim1 = 1
        g1[3] = 4096u << 16;             // tile_dim0 = 4096
        g1[4] = 0u;                      // tile_dim1/2 unused
        g1[5] = 4096u;                   // tensor_dim0_stride[31:0]
        g1[6] = 0u;
        g1[7] = 0u;
        asm volatile("tensor_load_to_lds %0, %1" :: "s"(g0), "s"(g1) : "memory");
        __builtin_amdgcn_s_wait_tensorcnt(0);
    }
    asm volatile("s_wait_asynccnt 0x0" ::: "memory");
    __syncthreads();

    const int wave = tid >> 5, lane = tid & 31;
    const int r = lane & 15, hi = lane >> 4;
    for (int mt = wave; mt < 16; mt += 4) {
        v8f acc = {};
        acc = wmma_tile<64>(At + mt * 16 * 64, 64, Bt, 64, acc, lane);
        float* out = xz + (size_t)(tok0 + r) * 256 + mt * 16 + hi * 8;
#pragma unroll
        for (int v = 0; v < 8; ++v) out[v] = acc[v];
    }
}

// ---------------------------------------------------------------------------
// Depthwise causal conv (k=4) + SiLU over xi = xz[:, :128]
// ---------------------------------------------------------------------------
__global__ void dwconv_kernel(const float* __restrict__ xz, const float* __restrict__ cw,
                              const float* __restrict__ cb,
                              float* __restrict__ xcf, _Float16* __restrict__ xch) {
    size_t i = (size_t)blockIdx.x * blockDim.x + threadIdx.x;   // BT*128
    int d = (int)(i & 127);
    size_t tok = i >> 7;
    int t = (int)(tok & 4095);
    float acc = cb[d];
#pragma unroll
    for (int k = 0; k < 4; ++k) {
        int ti = t - 3 + k;
        if (ti >= 0) acc += cw[d * 4 + k] * xz[(tok - 3 + k) * 256 + d];
    }
    float s = siluf(acc);
    xcf[i] = s;
    xch[i] = (_Float16)s;
}

// ---------------------------------------------------------------------------
// GEMM: xdb[tok, 0:36] = W_x(36x128, padded to 48) @ xc[tok]   (WMMA, K=128)
// Split epilogue into dt_r(4) | B(16) | C(16).
// ---------------------------------------------------------------------------
__global__ void gemm_x_kernel(const _Float16* __restrict__ xch,
                              const _Float16* __restrict__ WxH,
                              float* __restrict__ dtr, float* __restrict__ Bm,
                              float* __restrict__ Cm) {
    __shared__ _Float16 Bt[16 * 128];
    __shared__ _Float16 At[48 * 128];
    const int tid  = threadIdx.x;       // 128
    const int tok0 = blockIdx.x * 16;
    {
        const float4* src = (const float4*)(xch + (size_t)tok0 * 128);
        float4* dst = (float4*)Bt;
        dst[tid] = src[tid];
        dst[tid + 128] = src[tid + 128];
    }
    {
        const float4* src = (const float4*)WxH;     // 768 float4
        float4* dst = (float4*)At;
        for (int i = tid; i < 768; i += 128) dst[i] = src[i];
    }
    __syncthreads();
    const int wave = tid >> 5, lane = tid & 31;
    const int r = lane & 15, hi = lane >> 4;
    if (wave < 3) {
        v8f acc = {};
        acc = wmma_tile<128>(At + wave * 16 * 128, 128, Bt, 128, acc, lane);
        const size_t tok = (size_t)(tok0 + r);
#pragma unroll
        for (int v = 0; v < 8; ++v) {
            int ch = wave * 16 + hi * 8 + v;
            float val = acc[v];
            if (ch < 4)        dtr[tok * 4 + ch] = val;
            else if (ch < 20)  Bm[tok * 16 + (ch - 4)] = val;
            else if (ch < 36)  Cm[tok * 16 + (ch - 20)] = val;
        }
    }
}

// ---------------------------------------------------------------------------
// Chunked selective scan: h_t = exp(dt*A)*h + dt*B*x, y_t = <h,C> + D*x
// Pass 1: per-chunk partial state S and decay product P  (131072 threads)
// Pass 2: serial combine across the 64 chunks -> chunk start states
// Pass 3: replay chunk from start state, emit gated output
// ---------------------------------------------------------------------------
__global__ void scan_p1_kernel(const float* __restrict__ dtr, const float* __restrict__ Bm,
                               const float* __restrict__ xcf,
                               const float* __restrict__ Wdt, const float* __restrict__ bdt,
                               const float* __restrict__ Aneg,
                               float* __restrict__ Pbuf, float* __restrict__ Sbuf) {
    const int d = threadIdx.x;              // 128
    const int bc = blockIdx.x;              // b*64 + chunk
    const int b = bc >> 6, c = bc & 63;
    float a[16], h[16], P[16];
#pragma unroll
    for (int n = 0; n < 16; ++n) { a[n] = Aneg[d * 16 + n]; h[n] = 0.f; P[n] = 1.f; }
    const float w0 = Wdt[d * 4 + 0], w1 = Wdt[d * 4 + 1];
    const float w2 = Wdt[d * 4 + 2], w3 = Wdt[d * 4 + 3];
    const float bd = bdt[d];
    size_t tok = (size_t)b * 4096 + (size_t)c * 64;
    for (int t = 0; t < 64; ++t, ++tok) {
        float4 r4 = *(const float4*)(dtr + tok * 4);
        float dt = softplusf(w0 * r4.x + w1 * r4.y + w2 * r4.z + w3 * r4.w + bd);
        float x = xcf[tok * 128 + d];
        const float* Bp = Bm + tok * 16;
#pragma unroll
        for (int n = 0; n < 16; ++n) {
            float dA = __expf(dt * a[n]);
            h[n] = dA * h[n] + (dt * Bp[n]) * x;
            P[n] *= dA;
        }
    }
    float* Sp = Sbuf + ((size_t)bc * 128 + d) * 16;
    float* Pp = Pbuf + ((size_t)bc * 128 + d) * 16;
#pragma unroll
    for (int n = 0; n < 16; ++n) { Sp[n] = h[n]; Pp[n] = P[n]; }
}

__global__ void scan_p2_kernel(float* __restrict__ Pbuf, const float* __restrict__ Sbuf) {
    int i = blockIdx.x * blockDim.x + threadIdx.x;      // 16*128*16 = 32768
    int n = i & 15, d = (i >> 4) & 127, b = i >> 11;
    float H = 0.f;
    for (int c = 0; c < 64; ++c) {
        size_t idx = ((size_t)(b * 64 + c) * 128 + d) * 16 + n;
        float P = Pbuf[idx], S = Sbuf[idx];
        Pbuf[idx] = H;                  // chunk-start state for pass 3
        H = P * H + S;
    }
}

__global__ void scan_p3_kernel(const float* __restrict__ dtr, const float* __restrict__ Bm,
                               const float* __restrict__ Cm, const float* __restrict__ xcf,
                               const float* __restrict__ xz,
                               const float* __restrict__ Wdt, const float* __restrict__ bdt,
                               const float* __restrict__ Aneg, const float* __restrict__ Dp,
                               const float* __restrict__ Hs, _Float16* __restrict__ ygh) {
    const int d = threadIdx.x;
    const int bc = blockIdx.x;
    const int b = bc >> 6, c = bc & 63;
    float a[16], h[16];
    const float* Hp = Hs + ((size_t)bc * 128 + d) * 16;
#pragma unroll
    for (int n = 0; n < 16; ++n) { a[n] = Aneg[d * 16 + n]; h[n] = Hp[n]; }
    const float w0 = Wdt[d * 4 + 0], w1 = Wdt[d * 4 + 1];
    const float w2 = Wdt[d * 4 + 2], w3 = Wdt[d * 4 + 3];
    const float bd = bdt[d], Dd = Dp[d];
    size_t tok = (size_t)b * 4096 + (size_t)c * 64;
    for (int t = 0; t < 64; ++t, ++tok) {
        float4 r4 = *(const float4*)(dtr + tok * 4);
        float dt = softplusf(w0 * r4.x + w1 * r4.y + w2 * r4.z + w3 * r4.w + bd);
        float x = xcf[tok * 128 + d];
        float z = xz[tok * 256 + 128 + d];
        const float* Bp = Bm + tok * 16;
        const float* Cp = Cm + tok * 16;
        float y = 0.f;
#pragma unroll
        for (int n = 0; n < 16; ++n) {
            float dA = __expf(dt * a[n]);
            h[n] = dA * h[n] + (dt * Bp[n]) * x;
            y += h[n] * Cp[n];
        }
        float g = (y + x * Dd) * siluf(z);
        ygh[tok * 128 + d] = (_Float16)g;
    }
}

// ---------------------------------------------------------------------------
// GEMM: y += yg @ W_out^T (64x128)   (WMMA, K=128) -- residual, rewrites y/yh
// ---------------------------------------------------------------------------
__global__ void gemm_out_kernel(const _Float16* __restrict__ ygh,
                                const _Float16* __restrict__ WoutH,
                                float* __restrict__ y, _Float16* __restrict__ yh) {
    __shared__ _Float16 Bt[16 * 128];
    __shared__ _Float16 At[64 * 128];
    const int tid  = threadIdx.x;       // 128
    const int tok0 = blockIdx.x * 16;
    {
        const float4* src = (const float4*)(ygh + (size_t)tok0 * 128);
        float4* dst = (float4*)Bt;
        dst[tid] = src[tid];
        dst[tid + 128] = src[tid + 128];
    }
    {
        const float4* src = (const float4*)WoutH;   // 1024 float4
        float4* dst = (float4*)At;
        for (int i = tid; i < 1024; i += 128) dst[i] = src[i];
    }
    __syncthreads();
    const int wave = tid >> 5, lane = tid & 31;
    const int r = lane & 15, hi = lane >> 4;
    v8f acc = {};
    acc = wmma_tile<128>(At + wave * 16 * 128, 128, Bt, 128, acc, lane);
    size_t base = (size_t)(tok0 + r) * 64 + wave * 16 + hi * 8;
#pragma unroll
    for (int v = 0; v < 8; ++v) {
        float nv = y[base + v] + acc[v];
        y[base + v]  = nv;
        yh[base + v] = (_Float16)nv;
    }
}

// ---------------------------------------------------------------------------
// time flip / combine
// ---------------------------------------------------------------------------
__global__ void flip_kernel(const float* __restrict__ src, float* __restrict__ dst,
                            _Float16* __restrict__ dsth) {
    size_t i = (size_t)blockIdx.x * 256 + threadIdx.x;  // BT*64
    int c = (int)(i & 63);
    size_t tok = i >> 6;
    int t = (int)(tok & 4095);
    size_t b = tok >> 12;
    float v = src[(b * 4096 + (4095 - t)) * 64 + c];
    dst[i]  = v;
    dsth[i] = (_Float16)v;
}

__global__ void combine_kernel(float* __restrict__ y, const float* __restrict__ yrev,
                               _Float16* __restrict__ yh) {
    size_t i = (size_t)blockIdx.x * 256 + threadIdx.x;
    int c = (int)(i & 63);
    size_t tok = i >> 6;
    int t = (int)(tok & 4095);
    size_t b = tok >> 12;
    float v = 0.5f * (y[i] + yrev[(b * 4096 + (4095 - t)) * 64 + c]);
    y[i]  = v;
    yh[i] = (_Float16)v;
}

// ---------------------------------------------------------------------------
// Head: features = y @ Wf^T + bf   (WMMA, K=64) directly into d_out
// ---------------------------------------------------------------------------
__global__ void head_gemm_kernel(const _Float16* __restrict__ yh,
                                 const _Float16* __restrict__ WfH,
                                 const float* __restrict__ bf,
                                 float* __restrict__ feat) {
    __shared__ _Float16 Bt[16 * 64];
    __shared__ _Float16 At[64 * 64];
    const int tid  = threadIdx.x;       // 128
    const int tok0 = blockIdx.x * 16;
    ((float4*)Bt)[tid] = ((const float4*)(yh + (size_t)tok0 * 64))[tid];
    {
        const float4* src = (const float4*)WfH;     // 512 float4
        for (int i = tid; i < 512; i += 128) ((float4*)At)[i] = src[i];
    }
    __syncthreads();
    const int wave = tid >> 5, lane = tid & 31;
    const int r = lane & 15, hi = lane >> 4;
    v8f acc = {};
    acc = wmma_tile<64>(At + wave * 16 * 64, 64, Bt, 64, acc, lane);
    const int ch0 = wave * 16 + hi * 8;
    size_t base = (size_t)(tok0 + r) * 64 + ch0;
#pragma unroll
    for (int v = 0; v < 8; ++v) feat[base + v] = acc[v] + bf[ch0 + v];
}

__global__ void greduce_kernel(const float* __restrict__ feat, float* __restrict__ g) {
    __shared__ float sm[128];
    const int bc = blockIdx.x;          // b*64 + c
    const int b = bc >> 6, c = bc & 63;
    float s = 0.f;
    for (int t = threadIdx.x; t < 4096; t += 128)
        s += feat[((size_t)b * 4096 + t) * 64 + c];
    sm[threadIdx.x] = s;
    __syncthreads();
    for (int off = 64; off; off >>= 1) {
        if (threadIdx.x < off) sm[threadIdx.x] += sm[threadIdx.x + off];
        __syncthreads();
    }
    if (threadIdx.x == 0) g[b * 64 + c] = sm[0] * (1.f / 4096.f);
}

__global__ void head2_kernel(const float* __restrict__ g,
                             const float* __restrict__ Wq1, const float* __restrict__ bq1,
                             const float* __restrict__ Wq2, const float* __restrict__ bq2,
                             const float* __restrict__ We1, const float* __restrict__ be1,
                             const float* __restrict__ We2, const float* __restrict__ be2,
                             float* __restrict__ q, float* __restrict__ e) {
    int b = threadIdx.x;
    if (b >= 16) return;
    const float* gb = g + b * 64;
    float h1[32];
    for (int i = 0; i < 32; ++i) {
        float a = bq1[i];
        for (int k = 0; k < 64; ++k) a += Wq1[i * 64 + k] * gb[k];
        h1[i] = fmaxf(a, 0.f);
    }
    float ql = bq2[0];
    for (int i = 0; i < 32; ++i) ql += Wq2[i] * h1[i];
    q[b] = 1.f / (1.f + __expf(-ql));
    float h2[64];
    for (int i = 0; i < 64; ++i) {
        float a = be1[i];
        for (int k = 0; k < 64; ++k) a += We1[i * 64 + k] * gb[k];
        h2[i] = fmaxf(a, 0.f);
    }
    float lg[3];
    for (int j = 0; j < 3; ++j) {
        float a = be2[j];
        for (int i = 0; i < 64; ++i) a += We2[j * 64 + i] * h2[i];
        lg[j] = a;
    }
    float mx = fmaxf(lg[0], fmaxf(lg[1], lg[2]));
    float s0 = __expf(lg[0] - mx), s1 = __expf(lg[1] - mx), s2 = __expf(lg[2] - mx);
    float inv = 1.f / (s0 + s1 + s2);
    e[b * 3 + 0] = s0 * inv;
    e[b * 3 + 1] = s1 * inv;
    e[b * 3 + 2] = s2 * inv;
}

// ---------------------------------------------------------------------------
// host orchestration
// ---------------------------------------------------------------------------
extern "C" void kernel_launch(void* const* d_in, const int* in_sizes, int n_in,
                              void* d_out, int out_size, void* d_ws, size_t ws_size,
                              hipStream_t stream) {
    (void)in_sizes; (void)n_in; (void)out_size; (void)ws_size;
    constexpr size_t MB = 1024ull * 1024ull;
    constexpr int B = 16, T = 4096, BT = B * T;

    char* ws = (char*)d_ws;
    float*    Ya   = (float*)(ws + 0);            // BT*64 f32     16MB
    _Float16* Yha  = (_Float16*)(ws + 16 * MB);   // BT*64 f16      8MB
    float*    Yb   = (float*)(ws + 24 * MB);      // 16MB
    _Float16* Yhb  = (_Float16*)(ws + 40 * MB);   //  8MB
    float*    XZ   = (float*)(ws + 48 * MB);      // BT*256 f32    64MB
    float*    XCF  = (float*)(ws + 112 * MB);     // BT*128 f32    32MB
    _Float16* XCH  = (_Float16*)(ws + 144 * MB);  // BT*128 f16    16MB
    float*    DTR  = (float*)(ws + 160 * MB);     // BT*4           1MB
    float*    BM   = (float*)(ws + 161 * MB);     // BT*16          4MB
    float*    CM   = (float*)(ws + 165 * MB);     // BT*16          4MB
    _Float16* YGH  = (_Float16*)(ws + 169 * MB);  // BT*128 f16    16MB
    float*    SB   = (float*)(ws + 185 * MB);     // chunk states   8MB
    float*    PB   = (float*)(ws + 193 * MB);     // chunk decay    8MB
    char*     WH   = ws + 201 * MB;               // f16 weights
    _Float16* WfH  = (_Float16*)(ws + 201 * MB + 512 * 1024);
    float*    ANEG = (float*)(ws + 202 * MB);     // 4 * 8KB

    auto F = [&](int i) { return (const float*)d_in[i]; };
    // input layout (setup_inputs dict order, depth-first)
    // 0:x 1..6:conv1/bn1 7..12:conv2/bn2 13..18:conv3/bn3
    // 19+9L: W_in, conv_w, conv_b, W_x, W_dt, b_dt, A_log, D, W_out
    // 55:Wf 56:bf 57:Wq1 58:bq1 59:Wq2 60:bq2 61:We1 62:be1 63:We2 64:be2
    const int LBASE = 19, LSTRIDE = 9;

    _Float16* WinH[4]; _Float16* WxH[4]; _Float16* WoutH[4]; float* Aneg[4];
    for (int L = 0; L < 4; ++L) {
        char* base = WH + (size_t)L * 128 * 1024;
        WinH[L]  = (_Float16*)(base);                 // 32KB
        WxH[L]   = (_Float16*)(base + 32 * 1024);     // 12KB (48x128)
        WoutH[L] = (_Float16*)(base + 48 * 1024);     // 16KB
        Aneg[L]  = ANEG + (size_t)L * 2048;
        prep_layer_kernel<<<64, 256, 0, stream>>>(
            F(LBASE + L * LSTRIDE + 0), F(LBASE + L * LSTRIDE + 3),
            F(LBASE + L * LSTRIDE + 8), F(LBASE + L * LSTRIDE + 6),
            WinH[L], WxH[L], WoutH[L], Aneg[L]);
    }
    prep_wf_kernel<<<16, 256, 0, stream>>>(F(55), WfH);

    frontend_kernel<<<dim3(64, 16), 256, 0, stream>>>(
        F(0),
        F(1), F(2), F(3), F(4), F(5), F(6),
        F(7), F(8), F(9), F(10), F(11), F(12),
        F(13), F(14), F(15), F(16), F(17), F(18),
        Ya, Yha);

    auto mamba_pass = [&](float* Y, _Float16* Yh, int L) {
        const int lb = LBASE + L * LSTRIDE;
        gemm_in_kernel<<<BT / 16, 128, 0, stream>>>(Yh, WinH[L], XZ, BT);
        dwconv_kernel<<<BT * 128 / 256, 256, 0, stream>>>(XZ, F(lb + 1), F(lb + 2), XCF, XCH);
        gemm_x_kernel<<<BT / 16, 128, 0, stream>>>(XCH, WxH[L], DTR, BM, CM);
        scan_p1_kernel<<<B * 64, 128, 0, stream>>>(DTR, BM, XCF, F(lb + 4), F(lb + 5),
                                                   Aneg[L], PB, SB);
        scan_p2_kernel<<<128, 256, 0, stream>>>(PB, SB);
        scan_p3_kernel<<<B * 64, 128, 0, stream>>>(DTR, BM, CM, XCF, XZ, F(lb + 4),
                                                   F(lb + 5), Aneg[L], F(lb + 7), PB, YGH);
        gemm_out_kernel<<<BT / 16, 128, 0, stream>>>(YGH, WoutH[L], Y, Yh);
    };

    for (int L = 0; L < 4; ++L) mamba_pass(Ya, Yha, L);          // forward direction
    flip_kernel<<<BT * 64 / 256, 256, 0, stream>>>(Ya, Yb, Yhb); // reversed copy
    for (int L = 0; L < 4; ++L) mamba_pass(Yb, Yhb, L);          // reverse direction
    combine_kernel<<<BT * 64 / 256, 256, 0, stream>>>(Ya, Yb, Yha);

    float* feat = (float*)d_out;                          // BT*64
    float* qout = feat + (size_t)BT * 64;                 // 16
    float* eout = qout + 16;                              // 48
    float* gout = eout + 48;                              // 1024
    head_gemm_kernel<<<BT / 16, 128, 0, stream>>>(Yha, WfH, F(56), feat);
    greduce_kernel<<<B * 64, 128, 0, stream>>>(feat, gout);
    head2_kernel<<<1, 32, 0, stream>>>(gout, F(57), F(58), F(59), F(60),
                                       F(61), F(62), F(63), F(64), qout, eout);
}
